// Superpoint_MAE_55207509623408
// MI455X (gfx1250) — matrix operand order repaired and checked by reading, not compiled, not run on hardware
//
#include <hip/hip_runtime.h>
#include <hip/hip_bf16.h>
#include <math.h>

// ---------------- problem sizes ----------------
#define N_PTS   262144
#define S1_N    4096
#define S2_N    128
#define PAD_N   64
#define F_IN    11
#define H2_N    256
#define H3_N    512
#define C_N     384
#define PH_N    128

typedef __attribute__((ext_vector_type(16))) __bf16 v16bf;
typedef __attribute__((ext_vector_type(8)))  __bf16 v8bf;
typedef __attribute__((ext_vector_type(8)))  float  v8f;

// monotone float <-> uint key (order preserving), identity = key(-inf)
#define KEY_NEG_INF 0x007FFFFFu
__device__ __forceinline__ unsigned fkey(float x) {
    unsigned u = __float_as_uint(x);
    return (u & 0x80000000u) ? ~u : (u | 0x80000000u);
}
__device__ __forceinline__ float finv(unsigned k) {
    return (k & 0x80000000u) ? __uint_as_float(k ^ 0x80000000u) : __uint_as_float(~k);
}

__device__ __forceinline__ v8f wmma_bf16(v16bf a, v16bf b, v8f c) {
    return __builtin_amdgcn_wmma_f32_16x16x32_bf16(false, a, false, b, (short)0, c, false, false);
}
__device__ __forceinline__ v16bf load_frag(const __bf16* p0, const __bf16* p1) {
    union { v16bf v; v8bf h[2]; } u;
    u.h[0] = *(const v8bf*)p0;
    u.h[1] = *(const v8bf*)p1;
    return u.v;
}

// one K-step of the 2x2-fragment MMA (A from LDS at `astride`, B panel at stride 520)
__device__ __forceinline__ void mma_step(const __bf16* __restrict__ aBase, int astride,
                                         const __bf16* __restrict__ bPanel, int k,
                                         int wrOff, int wcOff, int nlane, int half8,
                                         v8f acc[2][2]) {
    v16bf afrag[2], bfrag[2];
#pragma unroll
    for (int fr = 0; fr < 2; ++fr) {
        const __bf16* base = aBase + (wrOff + fr * 16 + nlane) * astride;
        afrag[fr] = load_frag(base + half8, base + 16 + half8);
    }
#pragma unroll
    for (int fc = 0; fc < 2; ++fc) {
        const __bf16* base = bPanel + (wcOff + fc * 16 + nlane) * 520 + k;
        bfrag[fc] = load_frag(base + half8, base + 16 + half8);
    }
#pragma unroll
    for (int fr = 0; fr < 2; ++fr)
#pragma unroll
        for (int fc = 0; fc < 2; ++fc)
            acc[fr][fc] = wmma_bf16(afrag[fr], bfrag[fc], acc[fr][fc]);
}

// ---------------- workspace layout (bytes) ----------------
#define OFF_F2    ((size_t)0)            // N*256 bf16     = 134217728
#define OFF_GK    ((size_t)134217728)    // S1*256 u32     = 4194304
#define OFF_GBF   ((size_t)138412032)    // S1*256 bf16    = 2097152
#define OFF_TOKK  ((size_t)140509184)    // S1*384 u32     = 6291456
#define OFF_POS   ((size_t)146800640)    // S1*384 f32     = 6291456
#define OFF_W3T   ((size_t)153092096)    // 512*512 bf16   = 524288
#define OFF_W4T   ((size_t)153616384)    // 384*512 bf16   = 393216
#define OFF_WC    ((size_t)154009600)    // 11*256 f32     = 11264
#define OFF_BC    ((size_t)154020864)    // 256 f32        = 1024
#define OFF_RANK  ((size_t)154021888)    // S1 i32

// ---------------- tiny prep kernels ----------------
// Wc = W1@W2 [11,256], bc = b1@W2 + b2  (exact fold: no nonlinearity between them)
__global__ void wc_kernel(const float* __restrict__ W1, const float* __restrict__ b1,
                          const float* __restrict__ W2, const float* __restrict__ b2,
                          float* __restrict__ Wc, float* __restrict__ bc) {
    int c = threadIdx.x; // 256
    for (int k = 0; k < F_IN; ++k) {
        float a = 0.f;
        for (int j = 0; j < 128; ++j) a += W1[k * 128 + j] * W2[j * 256 + c];
        Wc[k * 256 + c] = a;
    }
    float a = b2[c];
    for (int j = 0; j < 128; ++j) a += b1[j] * W2[j * 256 + c];
    bc[c] = a;
}

// dst[n*K+k] = bf16(src[k*Nn+n])   (weights transposed to [n][k] for b128 panel staging)
__global__ void transpose_bf16_kernel(const float* __restrict__ src, __bf16* __restrict__ dst,
                                      int K, int Nn) {
    int idx = blockIdx.x * 256 + threadIdx.x;
    if (idx >= K * Nn) return;
    int n = idx / K, k = idx - n * K;
    dst[idx] = (__bf16)src[k * Nn + n];
}

__global__ void initkeys_kernel(unsigned* __restrict__ p, int n) {
    int i = blockIdx.x * 256 + threadIdx.x;
    if (i < n) p[i] = KEY_NEG_INF;
}
__global__ void zerof_kernel(float* __restrict__ p, int n) {
    int i = blockIdx.x * 256 + threadIdx.x;
    if (i < n) p[i] = 0.f;
}

// ---------------- embed: f2 = X@Wc+bc (bf16 out) + fused sorted segment-max ----------------
__global__ __launch_bounds__(256) void embed_kernel(const float* __restrict__ X,
                                                    const float* __restrict__ Wc,
                                                    const float* __restrict__ bc,
                                                    const int* __restrict__ idx10,
                                                    __bf16* __restrict__ f2_bf,
                                                    unsigned* __restrict__ gKeys) {
    __shared__ float xs[64 * F_IN];
    __shared__ int   segs[64];
    const int tid = threadIdx.x;       // == output column c (256)
    const int pb  = blockIdx.x * 64;   // point base
    for (int i = tid; i < 64 * F_IN; i += 256) xs[i] = X[(size_t)pb * F_IN + i];
    if (tid < 64) segs[tid] = idx10[pb + tid];
    float wcol[F_IN];
#pragma unroll
    for (int k = 0; k < F_IN; ++k) wcol[k] = Wc[k * 256 + tid];
    const float bcv = bc[tid];
    __syncthreads();

    int curSeg = -1; float curMax = 0.f;
    for (int p = 0; p < 64; ++p) {
        float v = bcv;
#pragma unroll
        for (int k = 0; k < F_IN; ++k) v += xs[p * F_IN + k] * wcol[k];
        f2_bf[(size_t)(pb + p) * 256 + tid] = (__bf16)v;
        int s = segs[p];
        if (s != curSeg) {
            if (curSeg >= 0) atomicMax(&gKeys[(size_t)curSeg * 256 + tid], fkey(curMax));
            curSeg = s; curMax = v;
        } else curMax = fmaxf(curMax, v);
    }
    atomicMax(&gKeys[(size_t)curSeg * 256 + tid], fkey(curMax));
}

__global__ void gconv_kernel(const unsigned* __restrict__ gKeys, __bf16* __restrict__ g_bf, int n) {
    int i = blockIdx.x * 256 + threadIdx.x;
    if (i < n) g_bf[i] = (__bf16)finv(gKeys[i]);
}

// ---------------- fused big-GEMM kernel ----------------
// Per block: 128 rows. Phase A: h1 = relu(concat(g[idx10], f2) @ W3 + b3) -> LDS (bf16, resident).
// Phase B: h2 = h1 @ W4 + b4, fused sorted-segment-max atomics into tokKeys.
#define FUSED_LDS 210432  // 512 idx + 10240 aTile(128x40) + 66560 bPanel(64x520) + 133120 h1(128x520)
__global__ __launch_bounds__(256) void fused_mlp_kernel(
        const __bf16* __restrict__ f2_bf, const __bf16* __restrict__ g_bf,
        const int* __restrict__ idx10,
        const __bf16* __restrict__ W3T, const float* __restrict__ b3,
        const __bf16* __restrict__ W4T, const float* __restrict__ b4,
        unsigned* __restrict__ tokKeys) {
    extern __shared__ char smem[];
    int*    idxCache = (int*)smem;                    // 128 ints
    __bf16* aTile    = (__bf16*)(smem + 512);         // [128][40]
    __bf16* bPanel   = (__bf16*)(smem + 10752);       // [64][520]  (weights, [n][k])
    __bf16* h1Buf    = (__bf16*)(smem + 77312);       // [128][520]

    const int tid   = threadIdx.x;
    const int lane  = tid & 31;
    const int wave  = tid >> 5;
    const int wrOff = (wave & 3) * 32;   // wave row offset (4 row-groups)
    const int wcOff = (wave >> 2) * 32;  // wave col offset (2 col-groups)
    const int nlane = lane & 15;
    const int half8 = (lane >> 4) * 8;   // 8*half
    const int rowbase = blockIdx.x * 128;

    if (tid < 128) idxCache[tid] = idx10[rowbase + tid];
    __syncthreads();

    // ---- hoisted A-tile staging pointers (branch-free inner loops) ----
    // thread stages two 8-element chunks per K-step: lin = tid and tid+256
    const int row0 = tid >> 2,         q0 = (tid & 3) * 8;
    const int row1 = (tid + 256) >> 2, q1 = ((tid + 256) & 3) * 8;
    const __bf16* gsrc0 = g_bf + (size_t)idxCache[row0] * 256 + q0;
    const __bf16* gsrc1 = g_bf + (size_t)idxCache[row1] * 256 + q1;
    const __bf16* fsrc0 = f2_bf + (size_t)(rowbase + row0) * 256 + q0;
    const __bf16* fsrc1 = f2_bf + (size_t)(rowbase + row1) * 256 + q1;
    __bf16* adst0 = aTile + row0 * 40 + q0;
    __bf16* adst1 = aTile + row1 * 40 + q1;

    // ---- hoisted segment ids for the Phase-B epilogue (constant across chunks) ----
    int segReg[16];
#pragma unroll
    for (int fr = 0; fr < 2; ++fr)
#pragma unroll
        for (int r = 0; r < 8; ++r)
            segReg[fr * 8 + r] = idxCache[wrOff + fr * 16 + r + half8];

    // ================= Phase A: h1 = relu(concat(g,f2) @ W3 + b3) =================
    for (int chunk = 0; chunk < 8; ++chunk) {
        const int cb = chunk * 64;
        // stage weight panel: 64 cols x 512 K (bf16, transposed layout) via b128 copies
#pragma unroll
        for (int i = 0; i < 16; ++i) {
            int lin = i * 256 + tid;
            int n = lin >> 6;
            int kc = (lin & 63) * 8;
            *(v8bf*)(bPanel + n * 520 + kc) = *(const v8bf*)(W3T + (size_t)(cb + n) * 512 + kc);
        }
        __syncthreads();

        v8f acc[2][2];
#pragma unroll
        for (int a0 = 0; a0 < 2; ++a0)
#pragma unroll
            for (int a1 = 0; a1 < 2; ++a1)
                acc[a0][a1] = (v8f){0.f, 0.f, 0.f, 0.f, 0.f, 0.f, 0.f, 0.f};

        // K in [0,256): gathered g_bf half (branch-free, pointers hoisted)
        for (int k0 = 0; k0 < 8; ++k0) {
            const int k = k0 * 32;
            *(v8bf*)adst0 = *(const v8bf*)(gsrc0 + k);
            *(v8bf*)adst1 = *(const v8bf*)(gsrc1 + k);
            __syncthreads();
            mma_step(aTile, 40, bPanel, k, wrOff, wcOff, nlane, half8, acc);
            __syncthreads();
        }
        // K in [256,512): f2_bf half
        for (int k0 = 8; k0 < 16; ++k0) {
            const int k = k0 * 32;
            *(v8bf*)adst0 = *(const v8bf*)(fsrc0 + (k - 256));
            *(v8bf*)adst1 = *(const v8bf*)(fsrc1 + (k - 256));
            __syncthreads();
            mma_step(aTile, 40, bPanel, k, wrOff, wcOff, nlane, half8, acc);
            __syncthreads();
        }

        // epilogue: bias + relu -> resident h1 (bf16). D layout: M=r+8*half, N=lane%16
#pragma unroll
        for (int fc = 0; fc < 2; ++fc) {
            const int col = wcOff + fc * 16 + nlane;
            const float bv = b3[cb + col];
#pragma unroll
            for (int fr = 0; fr < 2; ++fr) {
#pragma unroll
                for (int r = 0; r < 8; ++r) {
                    int row = wrOff + fr * 16 + r + half8;
                    float v = acc[fr][fc][r] + bv;
                    v = fmaxf(v, 0.f);
                    h1Buf[row * 520 + cb + col] = (__bf16)v;
                }
            }
        }
        __syncthreads();
    }

    // ================= Phase B: h2 = h1 @ W4 + b4, fused segment-max =================
    for (int chunk = 0; chunk < 6; ++chunk) {
        const int cb = chunk * 64;
#pragma unroll
        for (int i = 0; i < 16; ++i) {
            int lin = i * 256 + tid;
            int n = lin >> 6;
            int kc = (lin & 63) * 8;
            *(v8bf*)(bPanel + n * 520 + kc) = *(const v8bf*)(W4T + (size_t)(cb + n) * 512 + kc);
        }
        __syncthreads();

        v8f acc[2][2];
#pragma unroll
        for (int a0 = 0; a0 < 2; ++a0)
#pragma unroll
            for (int a1 = 0; a1 < 2; ++a1)
                acc[a0][a1] = (v8f){0.f, 0.f, 0.f, 0.f, 0.f, 0.f, 0.f, 0.f};

        // no barriers needed: both operands resident in LDS
#pragma unroll 4
        for (int k0 = 0; k0 < 16; ++k0)
            mma_step(h1Buf + k0 * 32, 520, bPanel, k0 * 32, wrOff, wcOff, nlane, half8, acc);

        // epilogue: bias + run-combined segment max atomics (idx10 sorted => few flushes)
#pragma unroll
        for (int fc = 0; fc < 2; ++fc) {
            const int gcol = cb + wcOff + fc * 16 + nlane;
            const float bv = b4[gcol];
            int curSeg = -1; float curMax = 0.f;
#pragma unroll
            for (int fr = 0; fr < 2; ++fr) {
#pragma unroll
                for (int r = 0; r < 8; ++r) {
                    float v = acc[fr][fc][r] + bv;
                    int s = segReg[fr * 8 + r];
                    if (s != curSeg) {
                        if (curSeg >= 0)
                            atomicMax(&tokKeys[(size_t)curSeg * C_N + gcol], fkey(curMax));
                        curSeg = s; curMax = v;
                    } else curMax = fmaxf(curMax, v);
                }
            }
            atomicMax(&tokKeys[(size_t)curSeg * C_N + gcol], fkey(curMax));
        }
        __syncthreads();
    }
}

// ---------------- pos embed: gelu(coords@P1+pb1)@P2+pb2 ----------------
__global__ __launch_bounds__(128) void pos_kernel(const float* __restrict__ coords,
                                                  const float* __restrict__ P1,
                                                  const float* __restrict__ pb1,
                                                  const float* __restrict__ P2,
                                                  const float* __restrict__ pb2,
                                                  float* __restrict__ pos) {
    __shared__ float hid[PH_N];
    const int row = blockIdx.x, t = threadIdx.x;
    const float x0 = coords[row * 3], x1 = coords[row * 3 + 1], x2 = coords[row * 3 + 2];
    float h = x0 * P1[t] + x1 * P1[PH_N + t] + x2 * P1[2 * PH_N + t] + pb1[t];
    h = 0.5f * h * (1.f + erff(h * 0.70710678118654752f));  // exact GELU
    hid[t] = h;
    __syncthreads();
    for (int cc = 0; cc < 3; ++cc) {
        int c = t + cc * PH_N;
        float a = pb2[c];
#pragma unroll 8
        for (int k = 0; k < PH_N; ++k) a += hid[k] * P2[k * C_N + c];
        pos[(size_t)row * C_N + c] = a;
    }
}

// ---------------- rank: s - lower_bound(idx21, idx21[s]) (idx21 sorted) ----------------
__global__ void rank_kernel(const int* __restrict__ idx21, int* __restrict__ rank) {
    int s = blockIdx.x * 256 + threadIdx.x;
    if (s >= S1_N) return;
    int v = idx21[s];
    int lo = 0, hi = S1_N;
    while (lo < hi) { int mid = (lo + hi) >> 1; if (idx21[mid] < v) lo = mid + 1; else hi = mid; }
    rank[s] = s - lo;
}

// ---------------- padded scatter (mode='drop') ----------------
__global__ __launch_bounds__(384) void scatter_kernel(const unsigned* __restrict__ tokKeys,
                                                      const float* __restrict__ pos,
                                                      const int* __restrict__ idx21,
                                                      const int* __restrict__ rank,
                                                      float* __restrict__ out) {
    const int s = blockIdx.x, c = threadIdx.x;
    const int r = rank[s];
    if (r >= PAD_N) return;
    const int g = idx21[s];
    const size_t base = ((size_t)g * PAD_N + r) * C_N + c;
    out[base] = finv(tokKeys[(size_t)s * C_N + c]);
    out[(size_t)S2_N * PAD_N * C_N + base] = pos[(size_t)s * C_N + c];
}

// ---------------- launcher ----------------
extern "C" void kernel_launch(void* const* d_in, const int* in_sizes, int n_in,
                              void* d_out, int out_size, void* d_ws, size_t ws_size,
                              hipStream_t stream) {
    (void)in_sizes; (void)n_in; (void)ws_size;
    const float* X      = (const float*)d_in[0];
    const float* coords = (const float*)d_in[1];
    const float* W1     = (const float*)d_in[2];
    const float* b1     = (const float*)d_in[3];
    const float* W2     = (const float*)d_in[4];
    const float* b2     = (const float*)d_in[5];
    const float* W3     = (const float*)d_in[6];
    const float* b3     = (const float*)d_in[7];
    const float* W4     = (const float*)d_in[8];
    const float* b4     = (const float*)d_in[9];
    const float* P1     = (const float*)d_in[10];
    const float* pb1    = (const float*)d_in[11];
    const float* P2     = (const float*)d_in[12];
    const float* pb2    = (const float*)d_in[13];
    const int*   idx10  = (const int*)d_in[14];
    const int*   idx21  = (const int*)d_in[15];
    float* out = (float*)d_out;

    char* ws = (char*)d_ws;
    __bf16*   f2_bf   = (__bf16*)(ws + OFF_F2);
    unsigned* gKeys   = (unsigned*)(ws + OFF_GK);
    __bf16*   g_bf    = (__bf16*)(ws + OFF_GBF);
    unsigned* tokKeys = (unsigned*)(ws + OFF_TOKK);
    float*    pos     = (float*)(ws + OFF_POS);
    __bf16*   W3T     = (__bf16*)(ws + OFF_W3T);
    __bf16*   W4T     = (__bf16*)(ws + OFF_W4T);
    float*    Wc      = (float*)(ws + OFF_WC);
    float*    bc      = (float*)(ws + OFF_BC);
    int*      rank    = (int*)(ws + OFF_RANK);

    hipFuncSetAttribute((const void*)fused_mlp_kernel,
                        hipFuncAttributeMaxDynamicSharedMemorySize, FUSED_LDS);

    // prep
    wc_kernel<<<1, 256, 0, stream>>>(W1, b1, W2, b2, Wc, bc);
    transpose_bf16_kernel<<<(H3_N * H3_N + 255) / 256, 256, 0, stream>>>(W3, W3T, H3_N, H3_N);
    transpose_bf16_kernel<<<(C_N * H3_N + 255) / 256, 256, 0, stream>>>(W4, W4T, H3_N, C_N);
    initkeys_kernel<<<(S1_N * H2_N + 255) / 256, 256, 0, stream>>>(gKeys, S1_N * H2_N);
    initkeys_kernel<<<(S1_N * C_N + 255) / 256, 256, 0, stream>>>(tokKeys, S1_N * C_N);

    // stage 1: folded embed + segment max
    embed_kernel<<<N_PTS / 64, 256, 0, stream>>>(X, Wc, bc, idx10, f2_bf, gKeys);
    gconv_kernel<<<(S1_N * H2_N + 255) / 256, 256, 0, stream>>>(gKeys, g_bf, S1_N * H2_N);

    // stage 2: fused WMMA double-GEMM with resident h1 + fused segment max
    fused_mlp_kernel<<<N_PTS / 128, 256, FUSED_LDS, stream>>>(
        f2_bf, g_bf, idx10, W3T, b3, W4T, b4, tokKeys);

    // stage 3: pos embed, ranks, padded scatter
    pos_kernel<<<S1_N, 128, 0, stream>>>(coords, P1, pb1, P2, pb2, pos);
    rank_kernel<<<(S1_N + 255) / 256, 256, 0, stream>>>(idx21, rank);
    zerof_kernel<<<(out_size + 255) / 256, 256, 0, stream>>>(out, out_size);
    scatter_kernel<<<S1_N, 384, 0, stream>>>(tokKeys, pos, idx21, rank, out);
}
